// CausalMultiHeadAttentionModule_55078660604405
// MI455X (gfx1250) — matrix operand order repaired and checked by reading, not compiled, not run on hardware
//
#include <hip/hip_runtime.h>
#include <hip/hip_bf16.h>

// Problem constants (reference: B=8, T=1024, E=512, H=8, D=64, WINDOW=128)
#define B_ 8
#define T_ 1024
#define E_ 512
#define H_ 8
#define D_ 64

typedef __attribute__((ext_vector_type(16))) _Float16 v16h;
typedef __attribute__((ext_vector_type(8)))  float    v8f;

union AF { unsigned u[8]; v16h v; _Float16 e[16]; };
union OF { float f[8]; v8f v; };

__device__ __forceinline__ v8f wmma16(v16h a, v16h b, v8f c) {
  // D = A(16x32 f16) * B(32x16 f16) + C(16x16 f32)
  return __builtin_amdgcn_wmma_f32_16x16x32_f16(false, a, false, b, (short)0, c,
                                                false, false);
}

// B-fragment (32x16, K-major per lane): lane n = L%16, kbase = (L<16?0:16),
// element h -> k = kbase + h. Data must be 16 contiguous halves at `p`.
__device__ __forceinline__ void load_bfrag(AF& b, const _Float16* p) {
  uint4 x0 = *reinterpret_cast<const uint4*>(p);
  uint4 x1 = *reinterpret_cast<const uint4*>(p + 8);
  b.u[0] = x0.x; b.u[1] = x0.y; b.u[2] = x0.z; b.u[3] = x0.w;
  b.u[4] = x1.x; b.u[5] = x1.y; b.u[6] = x1.z; b.u[7] = x1.w;
}

// A-fragment (16x32): lane m = L%16, kbase8 = (L<16?0:8),
// element h -> k = 2*kbase8 + h. `row` points at k=0 of this row.
__device__ __forceinline__ void load_afrag(AF& a, const _Float16* row, int kbase8) {
#pragma unroll
  for (int p = 0; p < 8; ++p) {
    int kk = (kbase8 << 1) + (p << 1);
    a.u[p] = *reinterpret_cast<const unsigned*>(row + kk);
  }
}

// ---------------------------------------------------------------- weights f32->f16
__global__ void cvt_f16_kernel(const float* __restrict__ src,
                               _Float16* __restrict__ dst, int n) {
  int i = blockIdx.x * blockDim.x + threadIdx.x;
  if (i < n) dst[i] = (_Float16)src[i];
}

// ---------------------------------------------------------------- LayerNorm (per row)
__global__ void ln_kernel(const float* __restrict__ x, const float* __restrict__ g,
                          const float* __restrict__ be, _Float16* __restrict__ xn) {
  const int row = blockIdx.x;                 // B_*T_ rows
  const float* xr = x + (size_t)row * E_;
  float s = 0.f, s2 = 0.f;
  for (int i = threadIdx.x; i < E_; i += blockDim.x) {
    float v = xr[i]; s += v; s2 += v * v;
  }
#pragma unroll
  for (int d = 1; d < 32; d <<= 1) {
    s  += __shfl_xor(s,  d, 32);
    s2 += __shfl_xor(s2, d, 32);
  }
  __shared__ float rs[8], rs2[8];
  const int wid = threadIdx.x >> 5;
  if ((threadIdx.x & 31) == 0) { rs[wid] = s; rs2[wid] = s2; }
  __syncthreads();
  float ts = 0.f, t2 = 0.f;
#pragma unroll
  for (int w = 0; w < 8; ++w) { ts += rs[w]; t2 += rs2[w]; }
  const float mu   = ts * (1.0f / E_);
  const float var  = t2 * (1.0f / E_) - mu * mu;
  const float rstd = rsqrtf(var + 1e-5f);
  _Float16* xo = xn + (size_t)row * E_;
  for (int i = threadIdx.x; i < E_; i += blockDim.x)
    xo[i] = (_Float16)((xr[i] - mu) * rstd * g[i] + be[i]);
}

// ---------------------------------------------------------------- QKV GEMM + routing
// A: xn [8192, 512] f16.  W: w_in [1536, 512] f16 (row n = output col, contiguous in k).
// Register-blocked: one wave -> 32x64 output (2 m-tiles x 4 n-tiles), 8 WMMA / k-step.
// grid = (8192/32, 24/8) = (256, 3), block = 256 (8 waves).
__global__ void qkv_gemm_kernel(const _Float16* __restrict__ A,
                                const _Float16* __restrict__ W,
                                const float* __restrict__ bias,
                                _Float16* __restrict__ qo,   // [BH, T, D], pre-scaled
                                _Float16* __restrict__ ko,   // [BH, T, D]
                                _Float16* __restrict__ vo) { // [BH, D, T] (transposed)
  const int lane = threadIdx.x & 31;
  const int wave = threadIdx.x >> 5;
  const int lm = lane & 15;
  const int hi = lane >> 4;
  const int mb = blockIdx.x;              // 32-row block
  const int nb = blockIdx.y * 8 + wave;   // 64-col block
  const _Float16* arow0 = A + (size_t)(mb * 32 + lm) * E_;
  const _Float16* arow1 = arow0 + (size_t)16 * E_;
  const _Float16* wbase = W + (size_t)(nb * 64 + lm) * E_ + (hi << 4);
  v8f c[2][4];
#pragma unroll
  for (int mi = 0; mi < 2; ++mi)
#pragma unroll
    for (int ni = 0; ni < 4; ++ni) c[mi][ni] = {};
#pragma unroll 2
  for (int k0 = 0; k0 < E_; k0 += 32) {
    AF a0, a1;
    load_afrag(a0, arow0 + k0, hi << 3);
    load_afrag(a1, arow1 + k0, hi << 3);
#pragma unroll
    for (int ni = 0; ni < 4; ++ni) {
      AF b;
      load_bfrag(b, wbase + (size_t)(ni * 16) * E_ + k0);
      c[0][ni] = wmma16(a0.v, b.v, c[0][ni]);
      c[1][ni] = wmma16(a1.v, b.v, c[1][ni]);
    }
  }
#pragma unroll
  for (int ni = 0; ni < 4; ++ni) {
    const int ng  = nb * 64 + ni * 16 + lm;
    const float bv = bias[ng];
    const int sel = ng >> 9;       // 0:q 1:k 2:v
    const int e   = ng & 511;
    const int hh  = e >> 6;
    const int dd  = e & 63;
#pragma unroll
    for (int mi = 0; mi < 2; ++mi) {
      OF cf; cf.v = c[mi][ni];
#pragma unroll
      for (int r = 0; r < 8; ++r) {
        const int mg = mb * 32 + mi * 16 + r + (hi << 3);
        const int bb = mg >> 10;
        const int tt = mg & 1023;
        const float val = cf.f[r] + bv;
        const size_t bh = (size_t)(bb * H_ + hh);
        if (sel == 0)      qo[(bh * T_ + tt) * D_ + dd] = (_Float16)(val * 0.125f); // 1/sqrt(64)
        else if (sel == 1) ko[(bh * T_ + tt) * D_ + dd] = (_Float16)val;
        else               vo[(bh * D_ + dd) * T_ + tt] = (_Float16)val;
      }
    }
  }
}

// ---------------------------------------------------------------- banded flash attention
// One wave per (bh, q-tile of 16).  grid = (T/16, B*H), block = 32.
__global__ void attn_kernel(const _Float16* __restrict__ Q,   // [BH,T,D]
                            const _Float16* __restrict__ K,   // [BH,T,D]
                            const _Float16* __restrict__ V,   // [BH,D,T]
                            _Float16* __restrict__ C) {       // ctx [B,T,E]
  __shared__ _Float16 pS[16 * 32];
  const int lane = threadIdx.x;
  const int lm   = lane & 15;
  const int hi   = lane >> 4;
  const int kb8  = hi << 3;
  const int kb16 = hi << 4;
  const int q0   = blockIdx.x * 16;
  const int bh   = blockIdx.y;
  const _Float16* Qb = Q + (size_t)bh * T_ * D_;
  const _Float16* Kb = K + (size_t)bh * T_ * D_;
  const _Float16* Vb = V + (size_t)bh * D_ * T_;
  const float NEGINF = -__builtin_inff();

  AF qa[2];  // Q A-fragments for D slices [0,32) and [32,64)
  {
    const _Float16* qrow = Qb + (size_t)(q0 + lm) * D_;
    load_afrag(qa[0], qrow, kb8);
    load_afrag(qa[1], qrow + 32, kb8);
  }

  float mrun[8], lrun[8], sc8[8];
  OF o[4];
#pragma unroll
  for (int r = 0; r < 8; ++r) { mrun[r] = NEGINF; lrun[r] = 0.f; }
#pragma unroll
  for (int t = 0; t < 4; ++t)
#pragma unroll
    for (int r = 0; r < 8; ++r) o[t].f[r] = 0.f;

  int ks = q0 - 128; if (ks < 0) ks = 0; ks &= ~31;
  for (int kp = ks; kp < q0 + 16; kp += 32) {
    // ---- scores S = Q * K^T for keys [kp, kp+32) : two 16x16 tiles
    OF s0, s1;
    {
      v8f acc = {};
      AF bf;
      const _Float16* kr = Kb + (size_t)(kp + lm) * D_ + kb16;
      load_bfrag(bf, kr);       acc = wmma16(qa[0].v, bf.v, acc);
      load_bfrag(bf, kr + 32);  acc = wmma16(qa[1].v, bf.v, acc);
      s0.v = acc;
    }
    {
      v8f acc = {};
      AF bf;
      const _Float16* kr = Kb + (size_t)(kp + 16 + lm) * D_ + kb16;
      load_bfrag(bf, kr);       acc = wmma16(qa[0].v, bf.v, acc);
      load_bfrag(bf, kr + 32);  acc = wmma16(qa[1].v, bf.v, acc);
      s1.v = acc;
    }
    // ---- banded-causal mask + online softmax (rows live in 16-lane groups)
    const int j0 = kp + lm;
    const int j1 = kp + 16 + lm;
#pragma unroll
    for (int r = 0; r < 8; ++r) {
      const int i = q0 + r + kb8;
      float a = s0.f[r]; if (j0 > i || j0 < i - 128) a = NEGINF;
      float b = s1.f[r]; if (j1 > i || j1 < i - 128) b = NEGINF;
      float mx = fmaxf(a, b);
#pragma unroll
      for (int d2 = 1; d2 < 16; d2 <<= 1) mx = fmaxf(mx, __shfl_xor(mx, d2, 32));
      const float mnew = fmaxf(mrun[r], mx);
      const bool dead  = mnew < -1e37f;          // all keys so far masked
      const float mexp = dead ? 0.f : mnew;
      const float scl  = dead ? 1.f : __expf(mrun[r] - mnew);
      const float p0 = __expf(a - mexp);
      const float p1 = __expf(b - mexp);
      float rs = p0 + p1;
#pragma unroll
      for (int d2 = 1; d2 < 16; d2 <<= 1) rs += __shfl_xor(rs, d2, 32);
      lrun[r] = lrun[r] * scl + rs;
      mrun[r] = mnew;
      sc8[r]  = scl;
      // stage P (16x32) into LDS in row-major for A-fragment reload
      pS[(r + kb8) * 32 + lm]      = (_Float16)p0;
      pS[(r + kb8) * 32 + 16 + lm] = (_Float16)p1;
    }
    asm volatile("s_wait_dscnt 0x0" ::: "memory");  // cross-lane LDS visibility
    AF pa;
#pragma unroll
    for (int p = 0; p < 8; ++p) {
      const int kk = kb16 + (p << 1);
      pa.u[p] = *reinterpret_cast<const unsigned*>(&pS[lm * 32 + kk]);
    }
    // ---- O = O * scale + P @ V  (4 d-tiles of 16)
#pragma unroll
    for (int t = 0; t < 4; ++t) {
#pragma unroll
      for (int r = 0; r < 8; ++r) o[t].f[r] *= sc8[r];
      AF vf;
      const _Float16* vr = Vb + (size_t)(t * 16 + lm) * T_ + kp + kb16;
      load_bfrag(vf, vr);
      o[t].v = wmma16(pa.v, vf.v, o[t].v);
    }
  }
  // ---- finalize: ctx[b, t, h*64+d] = O / l   (j == i always allowed -> lrun > 0)
  const int b  = bh >> 3;
  const int hh = bh & 7;
#pragma unroll
  for (int t = 0; t < 4; ++t)
#pragma unroll
    for (int r = 0; r < 8; ++r) {
      const int tg = q0 + r + kb8;
      const float val = o[t].f[r] / lrun[r];
      C[((size_t)(b * T_ + tg)) * E_ + hh * D_ + t * 16 + lm] = (_Float16)val;
    }
}

// ---------------------------------------------------------------- output projection
// ctx [8192,512] f16 x w_out[512,512] f16 -> out f32.
// Register-blocked 32x64 per wave.  grid = (256, 1), block = 256 (8 waves cover N=512).
__global__ void out_gemm_kernel(const _Float16* __restrict__ A,
                                const _Float16* __restrict__ W,
                                const float* __restrict__ bias,
                                float* __restrict__ out) {
  const int lane = threadIdx.x & 31;
  const int wave = threadIdx.x >> 5;
  const int lm = lane & 15;
  const int hi = lane >> 4;
  const int mb = blockIdx.x;      // 32-row block
  const int nb = wave;            // 64-col block (8 waves x 64 = 512)
  const _Float16* arow0 = A + (size_t)(mb * 32 + lm) * E_;
  const _Float16* arow1 = arow0 + (size_t)16 * E_;
  const _Float16* wbase = W + (size_t)(nb * 64 + lm) * E_ + (hi << 4);
  v8f c[2][4];
#pragma unroll
  for (int mi = 0; mi < 2; ++mi)
#pragma unroll
    for (int ni = 0; ni < 4; ++ni) c[mi][ni] = {};
#pragma unroll 2
  for (int k0 = 0; k0 < E_; k0 += 32) {
    AF a0, a1;
    load_afrag(a0, arow0 + k0, hi << 3);
    load_afrag(a1, arow1 + k0, hi << 3);
#pragma unroll
    for (int ni = 0; ni < 4; ++ni) {
      AF b;
      load_bfrag(b, wbase + (size_t)(ni * 16) * E_ + k0);
      c[0][ni] = wmma16(a0.v, b.v, c[0][ni]);
      c[1][ni] = wmma16(a1.v, b.v, c[1][ni]);
    }
  }
#pragma unroll
  for (int ni = 0; ni < 4; ++ni) {
    const int ng = nb * 64 + ni * 16 + lm;
    const float bv = bias[ng];
#pragma unroll
    for (int mi = 0; mi < 2; ++mi) {
      OF cf; cf.v = c[mi][ni];
#pragma unroll
      for (int r = 0; r < 8; ++r) {
        const int mg = mb * 32 + mi * 16 + r + (hi << 3);
        out[(size_t)mg * E_ + ng] = cf.f[r] + bv;
      }
    }
  }
}

extern "C" void kernel_launch(void* const* d_in, const int* in_sizes, int n_in,
                              void* d_out, int out_size, void* d_ws, size_t ws_size,
                              hipStream_t stream) {
  (void)in_sizes; (void)n_in; (void)out_size; (void)ws_size;
  const float* x     = (const float*)d_in[0];
  // d_in[1] = x_lengths (unused by the reference forward)
  const float* gamma = (const float*)d_in[2];
  const float* beta  = (const float*)d_in[3];
  const float* w_in  = (const float*)d_in[4];
  const float* b_in  = (const float*)d_in[5];
  const float* w_out = (const float*)d_in[6];
  const float* b_out = (const float*)d_in[7];
  float* out = (float*)d_out;

  char* ws = (char*)d_ws;
  size_t off = 0;
  auto take = [&](size_t bytes) -> void* {
    void* p = ws + off;
    off = (off + bytes + 255) & ~(size_t)255;
    return p;
  };
  _Float16* w_in_h  = (_Float16*)take((size_t)3 * E_ * E_ * 2);          // 1.5 MB
  _Float16* w_out_h = (_Float16*)take((size_t)E_ * E_ * 2);              // 0.5 MB
  _Float16* xn      = (_Float16*)take((size_t)B_ * T_ * E_ * 2);         // 8 MB
  _Float16* qh      = (_Float16*)take((size_t)B_ * H_ * T_ * D_ * 2);    // 8 MB
  _Float16* kh      = (_Float16*)take((size_t)B_ * H_ * T_ * D_ * 2);    // 8 MB
  _Float16* vh      = (_Float16*)take((size_t)B_ * H_ * T_ * D_ * 2);    // 8 MB
  _Float16* ctx     = (_Float16*)take((size_t)B_ * T_ * E_ * 2);         // 8 MB

  const int nwi = 3 * E_ * E_, nwo = E_ * E_;
  cvt_f16_kernel<<<(nwi + 255) / 256, 256, 0, stream>>>(w_in, w_in_h, nwi);
  cvt_f16_kernel<<<(nwo + 255) / 256, 256, 0, stream>>>(w_out, w_out_h, nwo);

  ln_kernel<<<B_ * T_, 256, 0, stream>>>(x, gamma, beta, xn);

  // M=8192 -> 256 m-blocks of 32; N=1536 -> 24 n-blocks of 64, 8 waves/block
  qkv_gemm_kernel<<<dim3(256, 3), 256, 0, stream>>>(xn, w_in_h, b_in, qh, kh, vh);

  attn_kernel<<<dim3(T_ / 16, B_ * H_), 32, 0, stream>>>(qh, kh, vh, ctx);

  // M=8192 -> 256 m-blocks; N=512 -> 8 n-blocks of 64 (one per wave)
  out_gemm_kernel<<<dim3(256, 1), 256, 0, stream>>>(ctx, w_out_h, b_out, out);
}
